// LGFIP_15564961480765
// MI455X (gfx1250) — matrix operand order repaired and checked by reading, not compiled, not run on hardware
//
#include <hip/hip_runtime.h>
#include <hip/hip_bf16.h>

// ---------------------------------------------------------------------------
// EdgeNeXt XCA block for MI455X (gfx1250, wave32).
// Heavy math (qkv / attn / proj / 3x3 conv) on v_wmma_f32_16x16x32_f16.
// v3: 16x64 wave tiles (no VGPR spills), ping-pong double buffering
//     (unroll-by-2, no rotation copies), tap-major conv loop.
// ---------------------------------------------------------------------------

typedef __attribute__((ext_vector_type(16))) _Float16 v16h;
typedef __attribute__((ext_vector_type(8)))  _Float16 v8h;
typedef __attribute__((ext_vector_type(8)))  float    v8f;

#define DIMC 384
#define NH   6
#define HD   64
#define NB   16
#define HH   56
#define WW   56
#define NSP  3136        // 56*56
#define NROWS 50176      // 16*3136

__device__ inline v8f zero8() {
  v8f z;
#pragma unroll
  for (int i = 0; i < 8; ++i) z[i] = 0.0f;
  return z;
}

__device__ inline v8f wmma16(v16h a, v16h b, v8f c) {
  return __builtin_amdgcn_wmma_f32_16x16x32_f16(false, a, false, b, (short)0, c,
                                                false, false);
}

// A fragment, row-major source, contiguous K order matching B:
// lane L (M = L&15): element e holds K = kb + e (kb = (L>>4)*16).
// Two contiguous 16B loads.
__device__ inline v16h frag_a(const _Float16* A, int lda, int lane) {
  int m  = lane & 15;
  int kb = (lane >> 4) << 4;
  const _Float16* p = A + (size_t)m * lda + kb;
  v8h lo = *(const v8h*)(p);
  v8h hi = *(const v8h*)(p + 8);
  v16h r;
#pragma unroll
  for (int i = 0; i < 8; ++i) { r[i] = lo[i]; r[i + 8] = hi[i]; }
  return r;
}

// B fragment from column-major storage: elem(k, col) at Bm[col*ld + k].
// B layout used here (16-bit, 32x16, contiguous K like frag_a): lanes 0-15 K=0..15, lanes 16-31 K=16..31,
// element e -> K = (lane>=16 ? 16 : 0) + e  => one contiguous 32B load.
__device__ inline v16h frag_b_cm(const _Float16* Bm, int ld, int lane) {
  int col = lane & 15;
  int kb  = (lane >> 4) << 4;
  return *(const v16h*)(Bm + (size_t)col * ld + kb);
}

// Conv A fragment for one tap: rowp points at the (b, hi, wj, 0) NHWC row,
// predicate `ok` handles zero padding, c0 = cs*32 within the 384 channels.
__device__ inline v16h frag_a_tap(const _Float16* rowp, bool ok, int c0, int lane) {
  int kb = (lane >> 4) << 4;
  v8h lo, hi;
#pragma unroll
  for (int i = 0; i < 8; ++i) { lo[i] = (_Float16)0.0f; hi[i] = (_Float16)0.0f; }
  if (ok) {
    lo = *(const v8h*)(rowp + c0 + kb);
    hi = *(const v8h*)(rowp + c0 + kb + 8);
  }
  v16h r;
#pragma unroll
  for (int i = 0; i < 8; ++i) { r[i] = lo[i]; r[i + 8] = hi[i]; }
  return r;
}

// ---------------------------------------------------------------------------
// 1. Fourier positional encoding -> projected (N, 384)
// ---------------------------------------------------------------------------
__global__ __launch_bounds__(384) void pos_kernel(const float* __restrict__ pos_w,
                                                  const float* __restrict__ pos_b,
                                                  float* __restrict__ pos_nc) {
  __shared__ float f[64];
  int n = blockIdx.x;
  int i = n / WW, j = n - i * WW;
  int t = threadIdx.x;
  if (t < 64) {
    const float scale = 6.283185307179586f;
    float e = (t < 32) ? ((float)(i + 1) / (56.0f + 1e-6f) * scale)
                       : ((float)(j + 1) / (56.0f + 1e-6f) * scale);
    int k = t & 31;
    float div = powf(10000.0f, (float)(2 * (k >> 1)) / 32.0f);
    float p = e / div;
    f[t] = (k & 1) ? cosf(p) : sinf(p);
  }
  __syncthreads();
  int c = t;  // blockDim == 384
  float acc = pos_b[c];
#pragma unroll
  for (int jj = 0; jj < 64; ++jj) acc += f[jj] * pos_w[c * 64 + jj];
  pos_nc[(size_t)n * DIMC + c] = acc;
}

// ---------------------------------------------------------------------------
// 2. NCHW -> (B,N,C) transpose + pos add (32x32 LDS tiles)
// ---------------------------------------------------------------------------
__global__ __launch_bounds__(256) void xpose_add(const float* __restrict__ x,
                                                 const float* __restrict__ pos_nc,
                                                 float* __restrict__ xt) {
  __shared__ float tile[32][33];
  int nt = blockIdx.x * 32, ct = blockIdx.y * 32, b = blockIdx.z;
  int tx = threadIdx.x, ty = threadIdx.y;
  const float* xb = x + (size_t)b * DIMC * NSP;
#pragma unroll
  for (int s = 0; s < 4; ++s)
    tile[ty + s * 8][tx] = xb[(size_t)(ct + ty + s * 8) * NSP + nt + tx];
  __syncthreads();
  float* xtb = xt + (size_t)b * NSP * DIMC;
#pragma unroll
  for (int s = 0; s < 4; ++s) {
    int n = nt + ty + s * 8;
    int c = ct + tx;
    xtb[(size_t)n * DIMC + c] = tile[tx][ty + s * 8] + pos_nc[(size_t)n * DIMC + c];
  }
}

// ---------------------------------------------------------------------------
// 3. LayerNorm over C=384 -> f16 activations (GEMM A operand)
// ---------------------------------------------------------------------------
__global__ __launch_bounds__(128) void ln_kernel(const float* __restrict__ xt,
                                                 const float* __restrict__ g,
                                                 const float* __restrict__ bb,
                                                 _Float16* __restrict__ xn) {
  __shared__ float red[128];
  int r = blockIdx.x, t = threadIdx.x;
  const float* row = xt + (size_t)r * DIMC;
  float v0 = row[t], v1 = row[t + 128], v2 = row[t + 256];
  red[t] = v0 + v1 + v2;
  __syncthreads();
  for (int o = 64; o > 0; o >>= 1) { if (t < o) red[t] += red[t + o]; __syncthreads(); }
  float mu = red[0] * (1.0f / 384.0f);
  __syncthreads();
  float d0 = v0 - mu, d1 = v1 - mu, d2 = v2 - mu;
  red[t] = d0 * d0 + d1 * d1 + d2 * d2;
  __syncthreads();
  for (int o = 64; o > 0; o >>= 1) { if (t < o) red[t] += red[t + o]; __syncthreads(); }
  float inv = rsqrtf(red[0] * (1.0f / 384.0f) + 1e-6f);
  _Float16* out = xn + (size_t)r * DIMC;
  out[t]       = (_Float16)(d0 * inv * g[t]       + bb[t]);
  out[t + 128] = (_Float16)(d1 * inv * g[t + 128] + bb[t + 128]);
  out[t + 256] = (_Float16)(d2 * inv * g[t + 256] + bb[t + 256]);
}

// ---------------------------------------------------------------------------
// 4. Weight packing into per-lane WMMA B-fragment layout
//    wp[((ks*ntiles+nt)*32 + lane)*16 + e] = W[o][k],
//    o = nt*16 + (lane&15), k = ks*32 + (lane>=16?16:0) + e
// ---------------------------------------------------------------------------
__global__ __launch_bounds__(256) void pack_dense(const float* __restrict__ W,
                                                  _Float16* __restrict__ wp,
                                                  int K, int O) {
  int idx = blockIdx.x * 256 + threadIdx.x;
  if (idx >= K * O) return;
  int e = idx & 15, lane = (idx >> 4) & 31, tt = idx >> 9;
  int ntiles = O >> 4;
  int ks = tt / ntiles, nt = tt - ks * ntiles;
  int o = (nt << 4) + (lane & 15);
  int k = (ks << 5) + ((lane >> 4) << 4) + e;
  wp[idx] = (_Float16)W[(size_t)o * K + k];
}

__global__ __launch_bounds__(256) void pack_conv(const float* __restrict__ W,
                                                 _Float16* __restrict__ wp) {
  // effective K = 9*384 with kk = (di*3+dj)*384 + c ; W is (O, C, 3, 3)
  int idx = blockIdx.x * 256 + threadIdx.x;
  if (idx >= 3456 * DIMC) return;
  int e = idx & 15, lane = (idx >> 4) & 31, tt = idx >> 9;
  const int ntiles = DIMC >> 4;  // 24
  int ks = tt / ntiles, nt = tt - ks * ntiles;
  int o = (nt << 4) + (lane & 15);
  int k = (ks << 5) + ((lane >> 4) << 4) + e;
  int tap = k / DIMC, c = k - tap * DIMC;
  int di = tap / 3, dj = tap - di * 3;
  wp[idx] = (_Float16)W[(((size_t)o * DIMC + c) * 3 + di) * 3 + dj];
}

// ---------------------------------------------------------------------------
// 5. qkv GEMM: (50176 x 384) x (384 x 1152), wave = 16x64 tile,
//    ping-pong double buffer (K unrolled by 2, no rotation copies).
//    scatter: q -> (b,h,d,N), k -> (b,h,d,N), v -> (b,h,N,d)
// ---------------------------------------------------------------------------
__global__ __launch_bounds__(256) void qkv_gemm(const _Float16* __restrict__ xn,
                                                const _Float16* __restrict__ wp,
                                                const float* __restrict__ qkv_b,
                                                _Float16* __restrict__ q_h,
                                                _Float16* __restrict__ k_h,
                                                _Float16* __restrict__ v_t) {
  int wid  = blockIdx.x * 8 + (threadIdx.x >> 5);
  int lane = threadIdx.x & 31;
  const int NG = 18;  // 1152 / 64
  int mt = wid / NG, ng = wid - mt * NG;
  int row0 = mt << 4;
  v8f acc[4];
#pragma unroll
  for (int t = 0; t < 4; ++t) acc[t] = zero8();
  const v16h* wv = (const v16h*)wp;
  const _Float16* a_base = xn + (size_t)row0 * DIMC;

  v16h aP = frag_a(a_base, DIMC, lane);
  v16h bP[4];
#pragma unroll
  for (int t = 0; t < 4; ++t) bP[t] = wv[(size_t)(ng * 4 + t) * 32 + lane];

#pragma unroll
  for (int ks = 0; ks < 12; ks += 2) {
    // load phase Q (ks+1, always < 12), compute phase P (ks)
    v16h aQ = frag_a(a_base + (ks + 1) * 32, DIMC, lane);
    v16h bQ[4];
#pragma unroll
    for (int t = 0; t < 4; ++t)
      bQ[t] = wv[((size_t)(ks + 1) * 72 + ng * 4 + t) * 32 + lane];
#pragma unroll
    for (int t = 0; t < 4; ++t) acc[t] = wmma16(aP, bP[t], acc[t]);
    // load phase P (ks+2, clamped), compute phase Q (ks+1)
    int k2 = (ks + 2 < 12) ? ks + 2 : ks;
    aP = frag_a(a_base + k2 * 32, DIMC, lane);
#pragma unroll
    for (int t = 0; t < 4; ++t)
      bP[t] = wv[((size_t)k2 * 72 + ng * 4 + t) * 32 + lane];
#pragma unroll
    for (int t = 0; t < 4; ++t) acc[t] = wmma16(aQ, bQ[t], acc[t]);
  }

  int col = lane & 15, half = lane >> 4;
  int o_base = ng * 64;
  int s = o_base / DIMC;
  int h = (o_base - s * DIMC) / HD;
  int b = row0 / NSP;
  int bh = b * NH + h;
  int n_base = row0 - b * NSP + 8 * half;
#pragma unroll
  for (int t = 0; t < 4; ++t) {
    int dd = t * 16 + col;
    float bias = qkv_b[o_base + dd];
    if (s == 0) {
      _Float16* dst = q_h + ((size_t)bh * HD + dd) * NSP + n_base;
#pragma unroll
      for (int r = 0; r < 8; ++r) dst[r] = (_Float16)(acc[t][r] + bias);
    } else if (s == 1) {
      _Float16* dst = k_h + ((size_t)bh * HD + dd) * NSP + n_base;
#pragma unroll
      for (int r = 0; r < 8; ++r) dst[r] = (_Float16)(acc[t][r] + bias);
    } else {
      _Float16* dst = v_t + ((size_t)bh * NSP + n_base) * HD + dd;
#pragma unroll
      for (int r = 0; r < 8; ++r) dst[(size_t)r * HD] = (_Float16)(acc[t][r] + bias);
    }
  }
}

// ---------------------------------------------------------------------------
// 6. reciprocal L2 norms of q/k rows over N (norm factored out of dot prod)
// ---------------------------------------------------------------------------
__global__ __launch_bounds__(256) void rownorm(const _Float16* __restrict__ q_h,
                                               const _Float16* __restrict__ k_h,
                                               float* __restrict__ qs,
                                               float* __restrict__ kss) {
  __shared__ float red[256];
  int r = blockIdx.x, t = threadIdx.x;
  const _Float16* base;
  float* out;
  if (r < NB * NH * HD) { base = q_h + (size_t)r * NSP; out = qs + r; }
  else { base = k_h + (size_t)(r - NB * NH * HD) * NSP; out = kss + (r - NB * NH * HD); }
  float s = 0.0f;
  for (int i = t; i < NSP; i += 256) { float v = (float)base[i]; s += v * v; }
  red[t] = s;
  __syncthreads();
  for (int o = 128; o > 0; o >>= 1) { if (t < o) red[t] += red[t + o]; __syncthreads(); }
  if (t == 0) *out = 1.0f / fmaxf(sqrtf(red[0]), 1e-12f);
}

// ---------------------------------------------------------------------------
// 7. per-(b,h): attn = softmax(qk^T * temp + bias), out = attn @ v -> xca f16
// ---------------------------------------------------------------------------
__global__ __launch_bounds__(512) void attn_kernel(const _Float16* __restrict__ q_h,
                                                   const _Float16* __restrict__ k_h,
                                                   const _Float16* __restrict__ v_t,
                                                   const float* __restrict__ qs,
                                                   const float* __restrict__ kss,
                                                   const float* __restrict__ temperature,
                                                   const float* __restrict__ rpb_table,
                                                   _Float16* __restrict__ xca) {
  __shared__ float attn_s[64][68];
  __shared__ __align__(16) _Float16 attn_h[64][64];
  int bh = blockIdx.x;
  int b = bh / NH, h = bh - b * NH;
  int w = threadIdx.x >> 5, lane = threadIdx.x & 31;
  const _Float16* qb = q_h + (size_t)bh * HD * NSP;
  const _Float16* kb = k_h + (size_t)bh * HD * NSP;
  {
    int mt = w >> 2, nt = w & 3;
    const _Float16* ap = qb + (size_t)mt * 16 * NSP;
    const _Float16* bp = kb + (size_t)(nt * 16) * NSP;
    v8f acc = zero8();
    v16h aP  = frag_a(ap, NSP, lane);
    v16h bfP = frag_b_cm(bp, NSP, lane);
#pragma unroll 1
    for (int ks = 0; ks < 98; ks += 2) {
      v16h aQ  = frag_a(ap + (ks + 1) * 32, NSP, lane);
      v16h bfQ = frag_b_cm(bp + (ks + 1) * 32, NSP, lane);
      acc = wmma16(aP, bfP, acc);
      int k2 = (ks + 2 < 98) ? ks + 2 : ks;
      aP  = frag_a(ap + k2 * 32, NSP, lane);
      bfP = frag_b_cm(bp + k2 * 32, NSP, lane);
      acc = wmma16(aQ, bfQ, acc);
    }
    float temp = temperature[h];
    float bias = rpb_table[1 * NH + h];  // rel_idx[0,0] == 1
    int col = lane & 15, half = lane >> 4;
    int n = nt * 16 + col;
    float kn2 = kss[bh * HD + n];
#pragma unroll
    for (int r = 0; r < 8; ++r) {
      int m = mt * 16 + 8 * half + r;
      attn_s[m][n] = acc[r] * qs[bh * HD + m] * kn2 * temp + bias;
    }
  }
  __syncthreads();
  if (threadIdx.x < 64) {
    int m = threadIdx.x;
    float mx = -1e30f;
#pragma unroll 4
    for (int e = 0; e < 64; ++e) mx = fmaxf(mx, attn_s[m][e]);
    float sum = 0.0f;
#pragma unroll 4
    for (int e = 0; e < 64; ++e) {
      float ev = __expf(attn_s[m][e] - mx);
      attn_s[m][e] = ev;
      sum += ev;
    }
    float inv = 1.0f / sum;
#pragma unroll 4
    for (int e = 0; e < 64; ++e) attn_h[m][e] = (_Float16)(attn_s[m][e] * inv);
  }
  __syncthreads();
  const _Float16* vb = v_t + (size_t)bh * NSP * HD;
  for (int t = w; t < 784; t += 16) {  // 4 m-tiles x 196 n-tiles
    int mt = t & 3, nt = t >> 2;
    v8f acc = zero8();
#pragma unroll
    for (int ks = 0; ks < 2; ++ks) {
      v16h a  = frag_a(&attn_h[mt * 16][ks * 32], 64, lane);
      v16h bf = frag_b_cm(vb + (size_t)(nt * 16) * HD + ks * 32, HD, lane);
      acc = wmma16(a, bf, acc);
    }
    int col = lane & 15, half = lane >> 4;
    int n_sp = nt * 16 + col;
    _Float16* dst = xca + ((size_t)b * NSP + n_sp) * DIMC + h * HD + mt * 16 + 8 * half;
#pragma unroll
    for (int r = 0; r < 8; ++r) dst[r] = (_Float16)acc[r];
  }
}

// ---------------------------------------------------------------------------
// 8. proj GEMM + gamma_xca residual -> x2 (NCHW f32) and x2 (NHWC f16)
//    wave = 16x64 tile, ping-pong double buffer
// ---------------------------------------------------------------------------
__global__ __launch_bounds__(256) void proj_gemm(const _Float16* __restrict__ xca,
                                                 const _Float16* __restrict__ wp,
                                                 const float* __restrict__ proj_b,
                                                 const float* __restrict__ gamma_xca,
                                                 const float* __restrict__ xt,
                                                 float* __restrict__ x2f,
                                                 _Float16* __restrict__ x2n) {
  int wid  = blockIdx.x * 8 + (threadIdx.x >> 5);
  int lane = threadIdx.x & 31;
  int mt = wid / 6, ng = wid - mt * 6;
  int row0 = mt << 4;
  v8f acc[4];
#pragma unroll
  for (int t = 0; t < 4; ++t) acc[t] = zero8();
  const v16h* wv = (const v16h*)wp;
  const _Float16* a_base = xca + (size_t)row0 * DIMC;

  v16h aP = frag_a(a_base, DIMC, lane);
  v16h bP[4];
#pragma unroll
  for (int t = 0; t < 4; ++t) bP[t] = wv[(size_t)(ng * 4 + t) * 32 + lane];

#pragma unroll
  for (int ks = 0; ks < 12; ks += 2) {
    v16h aQ = frag_a(a_base + (ks + 1) * 32, DIMC, lane);
    v16h bQ[4];
#pragma unroll
    for (int t = 0; t < 4; ++t)
      bQ[t] = wv[((size_t)(ks + 1) * 24 + ng * 4 + t) * 32 + lane];
#pragma unroll
    for (int t = 0; t < 4; ++t) acc[t] = wmma16(aP, bP[t], acc[t]);
    int k2 = (ks + 2 < 12) ? ks + 2 : ks;
    aP = frag_a(a_base + k2 * 32, DIMC, lane);
#pragma unroll
    for (int t = 0; t < 4; ++t)
      bP[t] = wv[((size_t)k2 * 24 + ng * 4 + t) * 32 + lane];
#pragma unroll
    for (int t = 0; t < 4; ++t) acc[t] = wmma16(aQ, bQ[t], acc[t]);
  }

  int col = lane & 15, half = lane >> 4;
  int b = row0 / NSP;
  int n_base = row0 - b * NSP + 8 * half;
#pragma unroll
  for (int t = 0; t < 4; ++t) {
    int c = ng * 64 + t * 16 + col;
    float pb = proj_b[c], gx = gamma_xca[c];
#pragma unroll
    for (int r = 0; r < 8; ++r) {
      int n_sp = n_base + r;
      size_t rrow = (size_t)b * NSP + n_sp;
      float val = xt[rrow * DIMC + c] + gx * (acc[t][r] + pb);
      x2f[((size_t)b * DIMC + c) * NSP + n_sp] = val;
      x2n[rrow * DIMC + c] = (_Float16)val;
    }
  }
}

// ---------------------------------------------------------------------------
// 9. 3x3 conv as implicit GEMM (K=3456) + BN + SiLU + residual -> x2_final
//    wave = 16x64 tile; tap-major loop, inner channel loop ping-ponged.
// ---------------------------------------------------------------------------
__global__ __launch_bounds__(256) void conv_gemm(const _Float16* __restrict__ x2n,
                                                 const _Float16* __restrict__ wp,
                                                 const float* __restrict__ bn_g,
                                                 const float* __restrict__ bn_b,
                                                 const float* __restrict__ bn_mean,
                                                 const float* __restrict__ bn_var,
                                                 const float* __restrict__ x2f,
                                                 float* __restrict__ x2fin) {
  int wid  = blockIdx.x * 8 + (threadIdx.x >> 5);
  int lane = threadIdx.x & 31;
  int mt = wid / 6, ng = wid - mt * 6;
  int row0 = mt << 4;
  int m0 = row0 + (lane & 15);
  int b  = m0 / NSP;                  // 16 | 3136: whole wave tile in one batch
  int r0 = m0 - b * NSP;
  int i0 = r0 / WW, j0 = r0 - i0 * WW;

  v8f acc[4];
#pragma unroll
  for (int t = 0; t < 4; ++t) acc[t] = zero8();
  const v16h* wv = (const v16h*)wp;

#pragma unroll 1
  for (int tap = 0; tap < 9; ++tap) {
    int di = tap / 3, dj = tap - di * 3;
    int h0 = i0 + di - 1, w0 = j0 + dj - 1;
    bool ok0 = (h0 >= 0) & (h0 < HH) & (w0 >= 0) & (w0 < WW);
    const _Float16* p0 = x2n + (((size_t)(b * HH + h0) * WW + w0) * DIMC);

    v16h aP = frag_a_tap(p0, ok0, 0, lane);
    v16h bP[4];
#pragma unroll
    for (int t = 0; t < 4; ++t)
      bP[t] = wv[((size_t)(tap * 12) * 24 + ng * 4 + t) * 32 + lane];

#pragma unroll
    for (int cs = 0; cs < 12; cs += 2) {
      v16h aQ = frag_a_tap(p0, ok0, (cs + 1) * 32, lane);
      v16h bQ[4];
#pragma unroll
      for (int t = 0; t < 4; ++t)
        bQ[t] = wv[((size_t)(tap * 12 + cs + 1) * 24 + ng * 4 + t) * 32 + lane];
#pragma unroll
      for (int t = 0; t < 4; ++t) acc[t] = wmma16(aP, bP[t], acc[t]);
      int c2 = (cs + 2 < 12) ? cs + 2 : cs;
      aP = frag_a_tap(p0, ok0, c2 * 32, lane);
#pragma unroll
      for (int t = 0; t < 4; ++t)
        bP[t] = wv[((size_t)(tap * 12 + c2) * 24 + ng * 4 + t) * 32 + lane];
#pragma unroll
      for (int t = 0; t < 4; ++t) acc[t] = wmma16(aQ, bQ[t], acc[t]);
    }
  }

  int col = lane & 15, half = lane >> 4;
  int n_base = row0 - b * NSP + 8 * half;
#pragma unroll
  for (int t = 0; t < 4; ++t) {
    int c = ng * 64 + t * 16 + col;
    float inv = rsqrtf(bn_var[c] + 1e-5f);
    float sc = inv * bn_g[c];
    float sh = bn_b[c] - bn_mean[c] * sc;
#pragma unroll
    for (int r = 0; r < 8; ++r) {
      float xi = acc[t][r] * sc + sh;
      float sil = xi / (1.0f + __expf(-xi));
      size_t idx = ((size_t)b * DIMC + c) * NSP + n_base + r;
      x2fin[idx] = x2f[idx] + sil;
    }
  }
}

// ---------------------------------------------------------------------------
// 10. final: faithful raw NCHW->(B,H,W,C) reshape * gamma, transpose, + x
// ---------------------------------------------------------------------------
__global__ __launch_bounds__(256) void final_kernel(const float* __restrict__ x,
                                                    const float* __restrict__ gamma,
                                                    const float* __restrict__ x2fin,
                                                    float* __restrict__ out) {
  size_t idx = (size_t)blockIdx.x * 256 + threadIdx.x;
  if (idx >= (size_t)NB * DIMC * NSP) return;
  int b  = (int)(idx / (DIMC * NSP));
  int w2 = (int)(idx - (size_t)b * DIMC * NSP);
  int c  = w2 / NSP;
  int p  = w2 - c * NSP;           // p = i*56 + j
  int f  = p * DIMC + c;           // flat index in (H,W,C) raw view
  int c_src = f / NSP;
  int rem   = f - c_src * NSP;
  out[idx] = x[idx] + gamma[c] * x2fin[(size_t)b * DIMC * NSP + (size_t)c_src * NSP + rem];
}

// ---------------------------------------------------------------------------
// launch
// ---------------------------------------------------------------------------
extern "C" void kernel_launch(void* const* d_in, const int* in_sizes, int n_in,
                              void* d_out, int out_size, void* d_ws, size_t ws_size,
                              hipStream_t stream) {
  const float* x         = (const float*)d_in[0];
  const float* pos_w     = (const float*)d_in[1];
  const float* pos_b     = (const float*)d_in[2];
  const float* ln_g      = (const float*)d_in[3];
  const float* ln_b      = (const float*)d_in[4];
  const float* gamma_xca = (const float*)d_in[5];
  const float* temperature = (const float*)d_in[6];
  const float* qkv_w     = (const float*)d_in[7];
  const float* qkv_b     = (const float*)d_in[8];
  const float* proj_w    = (const float*)d_in[9];
  const float* proj_b    = (const float*)d_in[10];
  const float* rpb_table = (const float*)d_in[11];
  const float* conv_w    = (const float*)d_in[12];
  const float* bn_g      = (const float*)d_in[13];
  const float* bn_b      = (const float*)d_in[14];
  const float* bn_mean   = (const float*)d_in[15];
  const float* bn_var    = (const float*)d_in[16];
  const float* gamma     = (const float*)d_in[17];

  // workspace layout (~302 MB), dead regions reused
  constexpr size_t SZ_F32 = (size_t)NROWS * DIMC * 4;  // 77,070,336
  constexpr size_t SZ_F16 = (size_t)NROWS * DIMC * 2;  // 38,535,168
  char* ws = (char*)d_ws;
  float*    xt    = (float*)(ws + 0);                       // later reused as x2fin
  _Float16* xn    = (_Float16*)(ws + SZ_F32);               // later reused as xca
  _Float16* q_h   = (_Float16*)(ws + SZ_F32 + SZ_F16);      // later reused as x2n
  _Float16* k_h   = (_Float16*)(ws + SZ_F32 + 2 * SZ_F16);
  _Float16* v_t   = (_Float16*)(ws + SZ_F32 + 3 * SZ_F16);
  float*    x2f   = (float*)(ws + SZ_F32 + 4 * SZ_F16);
  size_t off = 2 * SZ_F32 + 4 * SZ_F16;
  float*    pos_nc = (float*)(ws + off);              off += (size_t)NSP * DIMC * 4;
  _Float16* wqkv   = (_Float16*)(ws + off);           off += (size_t)DIMC * 1152 * 2;
  _Float16* wproj  = (_Float16*)(ws + off);           off += (size_t)DIMC * DIMC * 2;
  _Float16* wconv  = (_Float16*)(ws + off);           off += (size_t)3456 * DIMC * 2;
  float*    qs     = (float*)(ws + off);              off += (size_t)NB * NH * HD * 4;
  float*    kss    = (float*)(ws + off);
  _Float16* xca   = xn;
  _Float16* x2n   = q_h;
  float*    x2fin = xt;

  pos_kernel<<<NSP, DIMC, 0, stream>>>(pos_w, pos_b, pos_nc);
  xpose_add<<<dim3(NSP / 32, DIMC / 32, NB), dim3(32, 8), 0, stream>>>(x, pos_nc, xt);
  ln_kernel<<<NROWS, 128, 0, stream>>>(xt, ln_g, ln_b, xn);

  pack_dense<<<(DIMC * 1152 + 255) / 256, 256, 0, stream>>>(qkv_w, wqkv, DIMC, 1152);
  pack_dense<<<(DIMC * DIMC + 255) / 256, 256, 0, stream>>>(proj_w, wproj, DIMC, DIMC);
  pack_conv<<<(3456 * DIMC + 255) / 256, 256, 0, stream>>>(conv_w, wconv);

  qkv_gemm<<<(NROWS / 16) * 18 / 8, 256, 0, stream>>>(xn, wqkv, qkv_b, q_h, k_h, v_t);
  rownorm<<<2 * NB * NH * HD, 256, 0, stream>>>(q_h, k_h, qs, kss);
  attn_kernel<<<NB * NH, 512, 0, stream>>>(q_h, k_h, v_t, qs, kss, temperature,
                                           rpb_table, xca);
  proj_gemm<<<(NROWS / 16) * 6 / 8, 256, 0, stream>>>(xca, wproj, proj_b, gamma_xca,
                                                      xt, x2f, x2n);
  conv_gemm<<<(NROWS / 16) * 6 / 8, 256, 0, stream>>>(x2n, wconv, bn_g, bn_b,
                                                      bn_mean, bn_var, x2f, x2fin);
  final_kernel<<<(NB * DIMC * NSP + 255) / 256, 256, 0, stream>>>(x, gamma, x2fin,
                                                                  (float*)d_out);
  (void)in_sizes; (void)n_in; (void)out_size; (void)ws_size;
}
